// TransformerBlockQuantum_65481071408438
// MI455X (gfx1250) — matrix operand order repaired and checked by reading, not compiled, not run on hardware
//
#include <hip/hip_runtime.h>
#include <math.h>

// ---------------------------------------------------------------------------
// Quantum transformer block for MI455X (gfx1250, wave32, WMMA).
// - All GEMM-shaped math: v_wmma_f32_16x16x32_f16 (f16 in, f32 accumulate).
// - GEMMs: 4-wave workgroups, 64x64 tile, double-buffered async global->LDS
//   staging (ASYNCcnt path) + global prefetch of the tile-after-next.
// - Attention: flash-style (online softmax), no 201MB score tensor.
// ---------------------------------------------------------------------------

typedef _Float16 h16;
typedef __attribute__((ext_vector_type(16))) _Float16 v16h;
typedef __attribute__((ext_vector_type(8)))  float    v8f;

#define Bc  4
#define Sc  1024
#define Ec  768
#define Hc  12
#define Dc  64
#define FFc 3072
#define NQc 8
#define Mc  (Bc * Sc)   // 4096 rows

union FragU { v16h h; unsigned int u[8]; };

// A-matrix 16x32 f16 fragment: lane%16 = M row; lanes 0-15 hold
// K=0..15, lanes 16-31 hold K=16..31; VGPR r -> K pair 2r.
__device__ __forceinline__ v16h load_fragA(const h16* p, int ld) {
  const int lane = threadIdx.x & 31;
  const h16* row = p + (size_t)(lane & 15) * ld + ((lane >> 4) << 4);
  FragU f;
#pragma unroll
  for (int r = 0; r < 8; ++r) {
    const int k = 2 * r; // 0,2,4,...,14
    f.u[r] = *(const unsigned int*)(row + k);
  }
  return f.h;
}

// B-matrix 32x16 f16 fragment (contiguous k order, matching load_fragA): lane%16 = N column;
// lanes 0-15 hold K=0..15, lanes 16-31 hold K=16..31; VGPR r -> K pair 2r.
// p points at row n=0, k=0 of B^T stored [N,K] row-major.
__device__ __forceinline__ v16h load_fragB(const h16* p, int ld) {
  const int lane = threadIdx.x & 31;
  const h16* row = p + (size_t)(lane & 15) * ld + ((lane >> 4) << 4);
  FragU f;
#pragma unroll
  for (int r = 0; r < 8; ++r)
    f.u[r] = *(const unsigned int*)(row + 2 * r);
  return f.h;
}

// Same B layout, element (n,k) = p[n + k*kstride] (V tiles: n along d
// contiguous, k along sequence strided by E).
__device__ __forceinline__ v16h load_fragB_strided(const h16* p, int kstride) {
  const int lane = threadIdx.x & 31;
  const h16* base = p + (lane & 15) + (size_t)((lane >> 4) << 4) * kstride;
  FragU f;
#pragma unroll
  for (int r = 0; r < 8; ++r) {
    f.h[2 * r]     = base[(size_t)(2 * r) * kstride];
    f.h[2 * r + 1] = base[(size_t)(2 * r + 1) * kstride];
  }
  return f.h;
}

#define WMMA_F16(a, b, c) \
  __builtin_amdgcn_wmma_f32_16x16x32_f16(false, (a), false, (b), (short)0, (c), false, false)

// --- CDNA5 async global->LDS copy (ASYNCcnt) -------------------------------
// Per-lane: copies 16B from global addr to LDS offset.  Tracked by ASYNCcnt;
// async loads complete in order, so waiting <=4 releases the previous batch
// of 4 while the next batch is still in flight (double buffering).
__device__ __forceinline__ void async_copy_b128(unsigned lds_off, const void* gptr) {
  asm volatile("global_load_async_to_lds_b128 %0, %1, off"
               :: "v"(lds_off), "v"(gptr)
               : "memory");
}
__device__ __forceinline__ void wait_async_le4() {
  asm volatile("s_wait_asynccnt 0x4" ::: "memory");
}
__device__ __forceinline__ void wait_async_0() {
  asm volatile("s_wait_asynccnt 0x0" ::: "memory");
}

// ---------------------------------------------------------------------------
// GEMM: C[m,n] = sum_k A[m,k]*B[n,k]  (A:[M,K] f16, B:[N,K] f16, row-major).
// Block = 128 threads (4 waves) computing a 64x64 tile; each wave owns a
// 32x32 quadrant (2x2 WMMA tiles).  K-steps of 32 are staged through two LDS
// buffers (A 64x32 + B 64x32 = 8KB) by async copies: each thread issues 4
// b128 copies per step; step i+1's copies are in flight while step i computes.
// EPI=0: store f32.  EPI=1: store (h16)cos(acc + theta[n%64]) (fused quantum
// RX encode for the attention heads).
// ---------------------------------------------------------------------------
template <int EPI>
__global__ __launch_bounds__(128) void gemm_wmma_kernel(
    const h16* __restrict__ A, const h16* __restrict__ Bw, int Ndim, int K,
    const float* __restrict__ theta, float* __restrict__ Cf,
    h16* __restrict__ Ch) {
  __shared__ __align__(16) h16 Abuf[2][64 * 32];
  __shared__ __align__(16) h16 Bbuf[2][64 * 32];
  const int t = threadIdx.x;            // 0..127
  const int wave = t >> 5;
  const int n0 = blockIdx.x * 64;
  const int m0 = blockIdx.y * 64;
  const int wm = (wave >> 1) * 32;      // wave's quadrant inside the tile
  const int wn = (wave & 1) * 32;

  // Copy plan: 64x32 halfs = 256 x 16B chunks per matrix; thread t owns
  // chunks {t, t+128} of A and of B.  chunk c -> row c>>2, col (c&3)*8.
  const int r0 = t >> 2,           cc0 = (t & 3) * 8;
  const int r1 = (t + 128) >> 2,   cc1 = ((t + 128) & 3) * 8;

  auto issue = [&](int buf, int k) {
    unsigned abase = (unsigned)(uintptr_t)(&Abuf[buf][0]);
    unsigned bbase = (unsigned)(uintptr_t)(&Bbuf[buf][0]);
    async_copy_b128(abase + (unsigned)(r0 * 32 + cc0) * 2,
                    A + (size_t)(m0 + r0) * K + k + cc0);
    async_copy_b128(abase + (unsigned)(r1 * 32 + cc1) * 2,
                    A + (size_t)(m0 + r1) * K + k + cc1);
    async_copy_b128(bbase + (unsigned)(r0 * 32 + cc0) * 2,
                    Bw + (size_t)(n0 + r0) * K + k + cc0);
    async_copy_b128(bbase + (unsigned)(r1 * 32 + cc1) * 2,
                    Bw + (size_t)(n0 + r1) * K + k + cc1);
    if (k + 32 < K) {  // pull the tile-after-next toward L2
      __builtin_prefetch(A + (size_t)(m0 + r0) * K + k + 32 + cc0, 0, 1);
      __builtin_prefetch(Bw + (size_t)(n0 + r0) * K + k + 32 + cc0, 0, 1);
    }
  };

  v8f acc00 = {}, acc01 = {}, acc10 = {}, acc11 = {};
  issue(0, 0);
  int cur = 0;
  for (int k = 0; k < K; k += 32) {
    if (k + 32 < K) {
      issue(cur ^ 1, k + 32);   // 8 outstanding...
      wait_async_le4();         // ...release the 4 for buffer `cur`
    } else {
      wait_async_0();
    }
    __syncthreads();            // all waves' copies into `cur` visible
    const h16* As = &Abuf[cur][0];
    const h16* Bs = &Bbuf[cur][0];
    v16h a0 = load_fragA(As + (wm + 0) * 32, 32);
    v16h a1 = load_fragA(As + (wm + 16) * 32, 32);
    v16h b0 = load_fragB(Bs + (wn + 0) * 32, 32);
    v16h b1 = load_fragB(Bs + (wn + 16) * 32, 32);
    acc00 = WMMA_F16(a0, b0, acc00);
    acc01 = WMMA_F16(a0, b1, acc01);
    acc10 = WMMA_F16(a1, b0, acc10);
    acc11 = WMMA_F16(a1, b1, acc11);
    __syncthreads();            // reads of `cur` done before it is re-filled
    cur ^= 1;
  }

  const int lane = t & 31;
  const int colL = lane & 15, halfL = lane >> 4;
  const int mBase = m0 + wm, nBase = n0 + wn;
#pragma unroll
  for (int r = 0; r < 8; ++r) {
#define STORE_TILE(ACC, MI, NI)                                              \
    {                                                                        \
      int gm = mBase + (MI)*16 + r + 8 * halfL;                              \
      int gn = nBase + (NI)*16 + colL;                                       \
      float vv = (ACC)[r];                                                   \
      if (EPI == 1)                                                          \
        Ch[(size_t)gm * Ndim + gn] = (h16)cosf(vv + theta[gn & (Dc - 1)]);   \
      else                                                                   \
        Cf[(size_t)gm * Ndim + gn] = vv;                                     \
    }
    STORE_TILE(acc00, 0, 0)
    STORE_TILE(acc01, 0, 1)
    STORE_TILE(acc10, 1, 0)
    STORE_TILE(acc11, 1, 1)
#undef STORE_TILE
  }
}

// ---------------------------------------------------------------------------
// Flash attention: one wave per (b, h, 16-query tile).  Streams keys in
// 32-chunks: 4 WMMAs for scores (D=64 = 2 x K32), online softmax with 16-lane
// shfl reductions (rows live per-VGPR in the C layout), P re-layout via 1KB
// LDS into an A fragment, 4 WMMAs for O += P*V.  256 wmma / wave total.
// ---------------------------------------------------------------------------
__global__ __launch_bounds__(32) void flash_attn_kernel(
    const h16* __restrict__ qq, const h16* __restrict__ kq,
    const h16* __restrict__ vq, h16* __restrict__ attnv) {
  __shared__ __align__(16) h16 pLDS[16 * 32];
  const int b = blockIdx.z, h = blockIdx.y;
  const int q0 = blockIdx.x * 16;
  const int lane = threadIdx.x & 31;
  const int colL = lane & 15, halfL = lane >> 4;
  const size_t headoff = (size_t)h * Dc;

  const h16* qbase = qq + (size_t)(b * Sc + q0) * Ec + headoff;
  const v16h aq0 = load_fragA(qbase, Ec);       // d = 0..31
  const v16h aq1 = load_fragA(qbase + 32, Ec);  // d = 32..63

  v8f acc0 = {}, acc1 = {}, acc2 = {}, acc3 = {};
  float mrow[8], lrow[8];
#pragma unroll
  for (int r = 0; r < 8; ++r) { mrow[r] = -1e30f; lrow[r] = 0.f; }

  for (int kk = 0; kk < Sc; kk += 32) {
    const h16* kb = kq + (size_t)(b * Sc + kk) * Ec + headoff;
    v16h bk00 = load_fragB(kb, Ec);                       // keys kk..+15, d 0..31
    v16h bk01 = load_fragB(kb + 32, Ec);                  // keys kk..+15, d 32..63
    v16h bk10 = load_fragB(kb + (size_t)16 * Ec, Ec);     // keys kk+16..+31
    v16h bk11 = load_fragB(kb + (size_t)16 * Ec + 32, Ec);
    v8f s0 = {}, s1 = {};
    s0 = WMMA_F16(aq0, bk00, s0);
    s0 = WMMA_F16(aq1, bk01, s0);
    s1 = WMMA_F16(aq0, bk10, s1);
    s1 = WMMA_F16(aq1, bk11, s1);

#pragma unroll
    for (int r = 0; r < 8; ++r) {
      float sa = s0[r] * 0.125f;  // 1/sqrt(64)
      float sb = s1[r] * 0.125f;
      float mx = fmaxf(sa, sb);
#pragma unroll
      for (int off = 1; off < 16; off <<= 1)
        mx = fmaxf(mx, __shfl_xor(mx, off, 32));
      float mnew = fmaxf(mrow[r], mx);
      float corr = __expf(mrow[r] - mnew);
      float p0 = __expf(sa - mnew);
      float p1 = __expf(sb - mnew);
      float rs = p0 + p1;
#pragma unroll
      for (int off = 1; off < 16; off <<= 1)
        rs += __shfl_xor(rs, off, 32);
      lrow[r] = lrow[r] * corr + rs;
      mrow[r] = mnew;
      acc0[r] *= corr; acc1[r] *= corr; acc2[r] *= corr; acc3[r] *= corr;
      int rr = r + 8 * halfL;
      pLDS[rr * 32 + colL] = (h16)p0;
      pLDS[rr * 32 + colL + 16] = (h16)p1;
    }
    __syncthreads();
    v16h ap = load_fragA(pLDS, 32);  // P: 16 queries x 32 keys
    const h16* vb = vq + (size_t)(b * Sc + kk) * Ec + headoff;
    v16h bv0 = load_fragB_strided(vb + 0, Ec);
    v16h bv1 = load_fragB_strided(vb + 16, Ec);
    v16h bv2 = load_fragB_strided(vb + 32, Ec);
    v16h bv3 = load_fragB_strided(vb + 48, Ec);
    acc0 = WMMA_F16(ap, bv0, acc0);
    acc1 = WMMA_F16(ap, bv1, acc1);
    acc2 = WMMA_F16(ap, bv2, acc2);
    acc3 = WMMA_F16(ap, bv3, acc3);
    __syncthreads();
  }

  h16* obase = attnv + (size_t)(b * Sc + q0) * Ec + headoff;
#pragma unroll
  for (int r = 0; r < 8; ++r) {
    float inv = 1.f / lrow[r];
    int rr = r + 8 * halfL;
    h16* orow = obase + (size_t)rr * Ec;
    orow[0  + colL] = (h16)(acc0[r] * inv);
    orow[16 + colL] = (h16)(acc1[r] * inv);
    orow[32 + colL] = (h16)(acc2[r] * inv);
    orow[48 + colL] = (h16)(acc3[r] * inv);
  }
}

// ---------------------------------------------------------------------------
// Elementwise / small kernels
// ---------------------------------------------------------------------------
__global__ __launch_bounds__(256) void f32_to_f16_kernel(
    const float* __restrict__ s, h16* __restrict__ d, int n) {
  int i = blockIdx.x * 256 + threadIdx.x;
  if (i < n) d[i] = (h16)s[i];
}

// out = LayerNorm(a + b (+ bias)) * g + beta.   One block per token row.
__global__ __launch_bounds__(256) void ln_kernel(
    const float* __restrict__ a, const float* __restrict__ b,
    const float* __restrict__ bias, const float* __restrict__ g,
    const float* __restrict__ beta, float* __restrict__ outf) {
  __shared__ float red[256];
  const int row = blockIdx.x, t = threadIdx.x;
  const float* pa = a + (size_t)row * Ec;
  const float* pb = b + (size_t)row * Ec;
  float v[3];
  float s = 0.f;
#pragma unroll
  for (int i = 0; i < 3; ++i) {
    int e = t + i * 256;
    float xv = pa[e] + pb[e] + (bias ? bias[e] : 0.f);
    v[i] = xv;
    s += xv;
  }
  red[t] = s; __syncthreads();
  for (int o = 128; o > 0; o >>= 1) { if (t < o) red[t] += red[t + o]; __syncthreads(); }
  float mu = red[0] * (1.f / Ec);
  __syncthreads();
  float vs = 0.f;
#pragma unroll
  for (int i = 0; i < 3; ++i) { float dv = v[i] - mu; vs += dv * dv; }
  red[t] = vs; __syncthreads();
  for (int o = 128; o > 0; o >>= 1) { if (t < o) red[t] += red[t + o]; __syncthreads(); }
  float rstd = rsqrtf(red[0] * (1.f / Ec) + 1e-5f);
#pragma unroll
  for (int i = 0; i < 3; ++i) {
    int e = t + i * 256;
    outf[(size_t)row * Ec + e] = (v[i] - mu) * rstd * g[e] + beta[e];
  }
}

// proj = x1 @ in_proj_w^T + b ; qout = cos(theta)*cos(proj).  NQ=8 -> VALU.
__global__ __launch_bounds__(256) void inproj_kernel(
    const float* __restrict__ x1, const float* __restrict__ W,
    const float* __restrict__ bias, const float* __restrict__ theta,
    float* __restrict__ qout) {
  __shared__ float sums[NQc][8];
  const int m = blockIdx.x, t = threadIdx.x;
  const float* xr = x1 + (size_t)m * Ec;
  float part[NQc];
#pragma unroll
  for (int j = 0; j < NQc; ++j) part[j] = 0.f;
  for (int e = t; e < Ec; e += 256) {
    float xv = xr[e];
#pragma unroll
    for (int j = 0; j < NQc; ++j) part[j] += xv * W[(size_t)j * Ec + e];
  }
#pragma unroll
  for (int j = 0; j < NQc; ++j)
#pragma unroll
    for (int off = 1; off < 32; off <<= 1)
      part[j] += __shfl_xor(part[j], off, 32);
  const int w = t >> 5, l = t & 31;
  if (l == 0) {
#pragma unroll
    for (int j = 0; j < NQc; ++j) sums[j][w] = part[j];
  }
  __syncthreads();
  if (t < NQc) {
    float s = bias[t];
#pragma unroll
    for (int w2 = 0; w2 < 8; ++w2) s += sums[t][w2];
    qout[(size_t)m * NQc + t] = cosf(theta[t]) * cosf(s);
  }
}

// h = relu(qout @ W1^T + b1), stored f16 for the W2 WMMA GEMM.  K=8 -> VALU.
__global__ __launch_bounds__(256) void ffn1_kernel(
    const float* __restrict__ qout, const float* __restrict__ W1,
    const float* __restrict__ b1, h16* __restrict__ hbuf) {
  int idx = blockIdx.x * 256 + threadIdx.x;
  if (idx >= Mc * FFc) return;
  int m = idx / FFc, n = idx - m * FFc;
  float s = b1[n];
  const float* q = qout + (size_t)m * NQc;
  const float* w = W1 + (size_t)n * NQc;
#pragma unroll
  for (int j = 0; j < NQc; ++j) s += q[j] * w[j];
  hbuf[idx] = (h16)fmaxf(s, 0.f);
}

// ---------------------------------------------------------------------------
extern "C" void kernel_launch(void* const* d_in, const int* in_sizes, int n_in,
                              void* d_out, int out_size, void* d_ws,
                              size_t ws_size, hipStream_t stream) {
  (void)in_sizes; (void)n_in; (void)out_size;
  const float* x          = (const float*)d_in[0];
  const float* Wq         = (const float*)d_in[1];
  const float* Wk         = (const float*)d_in[2];
  const float* Wv         = (const float*)d_in[3];
  const float* Wc         = (const float*)d_in[4];
  const float* theta_attn = (const float*)d_in[5];
  const float* in_proj_w  = (const float*)d_in[6];
  const float* in_proj_b  = (const float*)d_in[7];
  const float* theta_ffn  = (const float*)d_in[8];
  const float* W1         = (const float*)d_in[9];
  const float* b1         = (const float*)d_in[10];
  const float* W2         = (const float*)d_in[11];
  const float* b2         = (const float*)d_in[12];
  const float* g1         = (const float*)d_in[13];
  const float* beta1      = (const float*)d_in[14];
  const float* g2         = (const float*)d_in[15];
  const float* beta2      = (const float*)d_in[16];

  char* ws = (char*)d_ws;
  size_t off = 0;
  auto alloc = [&](size_t bytes) -> char* {
    char* p = ws + off;
    off += (bytes + 255) & ~(size_t)255;
    return p;
  };
  h16*   xh       = (h16*)alloc((size_t)Mc * Ec * 2);
  h16*   wqh      = (h16*)alloc((size_t)Ec * Ec * 2);
  h16*   wkh      = (h16*)alloc((size_t)Ec * Ec * 2);
  h16*   wvh      = (h16*)alloc((size_t)Ec * Ec * 2);
  h16*   wch      = (h16*)alloc((size_t)Ec * Ec * 2);
  h16*   w2h      = (h16*)alloc((size_t)Ec * FFc * 2);
  // qq/kq/vq/attnv region is exactly Mc*FFc*2 bytes; reused later for h.
  char*  region   = alloc((size_t)Mc * FFc * 2);
  h16*   qqb      = (h16*)region;
  h16*   kqb      = qqb + (size_t)Mc * Ec;
  h16*   vqb      = kqb + (size_t)Mc * Ec;
  h16*   attnv    = vqb + (size_t)Mc * Ec;
  h16*   hbuf     = (h16*)region;                       // reuse after attention
  float* attn_out = (float*)alloc((size_t)Mc * Ec * 4);
  float* ffn_out  = attn_out;                           // reuse after LN1
  float* x1       = (float*)alloc((size_t)Mc * Ec * 4);
  float* qout     = (float*)alloc((size_t)Mc * NQc * 4);
  if (off > ws_size) return;  // workspace too small; bail deterministically

  // 1) f32 -> f16 operand conversion
  f32_to_f16_kernel<<<(Mc * Ec + 255) / 256, 256, 0, stream>>>(x, xh, Mc * Ec);
  f32_to_f16_kernel<<<(Ec * Ec + 255) / 256, 256, 0, stream>>>(Wq, wqh, Ec * Ec);
  f32_to_f16_kernel<<<(Ec * Ec + 255) / 256, 256, 0, stream>>>(Wk, wkh, Ec * Ec);
  f32_to_f16_kernel<<<(Ec * Ec + 255) / 256, 256, 0, stream>>>(Wv, wvh, Ec * Ec);
  f32_to_f16_kernel<<<(Ec * Ec + 255) / 256, 256, 0, stream>>>(Wc, wch, Ec * Ec);
  f32_to_f16_kernel<<<(Ec * FFc + 255) / 256, 256, 0, stream>>>(W2, w2h, Ec * FFc);

  // 2) QKV projections, fused cos(q + theta) epilogue -> f16 qq/kq/vq
  dim3 gE(Ec / 64, Mc / 64);
  gemm_wmma_kernel<1><<<gE, 128, 0, stream>>>(xh, wqh, Ec, Ec, theta_attn, nullptr, qqb);
  gemm_wmma_kernel<1><<<gE, 128, 0, stream>>>(xh, wkh, Ec, Ec, theta_attn, nullptr, kqb);
  gemm_wmma_kernel<1><<<gE, 128, 0, stream>>>(xh, wvh, Ec, Ec, theta_attn, nullptr, vqb);

  // 3) Flash attention
  dim3 gA(Sc / 16, Hc, Bc);
  flash_attn_kernel<<<gA, 32, 0, stream>>>(qqb, kqb, vqb, attnv);

  // 4) Output projection Wc
  gemm_wmma_kernel<0><<<gE, 128, 0, stream>>>(attnv, wch, Ec, Ec, nullptr, attn_out, nullptr);

  // 5) x1 = LN(x + attn_out)
  ln_kernel<<<Mc, 256, 0, stream>>>(x, attn_out, nullptr, g1, beta1, x1);

  // 6) quantum FFN front half (NQ=8, VALU)
  inproj_kernel<<<Mc, 256, 0, stream>>>(x1, in_proj_w, in_proj_b, theta_ffn, qout);
  ffn1_kernel<<<(Mc * FFc) / 256, 256, 0, stream>>>(qout, W1, b1, hbuf);

  // 7) W2 GEMM (K = 3072)
  gemm_wmma_kernel<0><<<gE, 128, 0, stream>>>(hbuf, w2h, Ec, FFc, nullptr, ffn_out, nullptr);

  // 8) out = LN(x1 + ffn_out + b2)
  ln_kernel<<<Mc, 256, 0, stream>>>(x1, ffn_out, b2, g2, beta2, (float*)d_out);
}
